// PacketRoutingModel_64802466562316
// MI455X (gfx1250) — matrix operand order repaired and checked by reading, not compiled, not run on hardware
//
#include <hip/hip_runtime.h>
#include <math.h>

typedef __attribute__((ext_vector_type(16))) _Float16 v16h;
typedef __attribute__((ext_vector_type(8)))  _Float16 v8h;
typedef __attribute__((ext_vector_type(4)))  _Float16 v4h;
typedef __attribute__((ext_vector_type(8)))  float    v8f;

#define HD   128
#define NTOK (64 * 2048)

// f16 weight workspace layout (offsets in halves, all 16B aligned)
#define WS_OBSW   0          // 128x64
#define WS_PREW1  8192       // 256x640
#define WS_PREW2  172032     // 128x256
#define WS_WIH    204800     // 384x128
#define WS_WHH    253952     // 384x128
#define WS_PKW1   303104     // 256x512
#define WS_PKW2   434176     // 128x256
#define WS_RTW    466944     // 128x512
#define WS_TOTAL  532480     // halves -> 1,064,960 bytes

struct Params {
  const float* packet; const float* node; const float* obs;
  const float* age; const float* timef; const float* rem;
  const int*   nidx;
  const float* obs_W;  const float* obs_b;
  const float* meta_W; const float* meta_b;
  const float* embed;
  const float* pre_g;  const float* pre_bln;
  const float* pre_W1; const float* pre_b1;
  const float* pre_W2; const float* pre_b2;
  const float* Wih; const float* Whh; const float* bih; const float* bhh;
  const float* pk_g;  const float* pk_bln;
  const float* pk_W1; const float* pk_b1;
  const float* pk_W2; const float* pk_b2;
  const float* ad_down; const float* ad_up;
  const float* rt_g;  const float* rt_bln;
  const float* rt_W;  const float* rt_b;
  const float* ro_W;  const float* ro_b;
  float* out_ns; float* out_pk; float* out_lg;
};

__device__ __forceinline__ float geluf(float x) {
  return 0.5f * x * (1.0f + erff(x * 0.70710678118654752f));
}

__device__ __forceinline__ void st8h(_Float16* dst, float4 a, float4 b) {
  v8h t;
  t[0] = (_Float16)a.x; t[1] = (_Float16)a.y; t[2] = (_Float16)a.z; t[3] = (_Float16)a.w;
  t[4] = (_Float16)b.x; t[5] = (_Float16)b.y; t[6] = (_Float16)b.z; t[7] = (_Float16)b.w;
  *reinterpret_cast<v8h*>(dst) = t;
}

// ---- WMMA helpers (16x16x32 f16 -> f32), wave32 ------------------------------
// A tile 16x32 f16 from LDS: lane L<16 -> row L, K {0..15};
// lane L>=16 -> row L-16, K {16..31}. Two ds_load_b128 per lane.
__device__ __forceinline__ v16h ldsA(const _Float16* X, int strideH, int k0) {
  const int lane = threadIdx.x & 31;
  const _Float16* p = X + (lane & 15) * strideH + k0 + ((lane >> 4) & 1) * 16;
  v16h a;
  *reinterpret_cast<uint4*>(&a)       = *reinterpret_cast<const uint4*>(p);
  *(reinterpret_cast<uint4*>(&a) + 1) = *reinterpret_cast<const uint4*>(p + 8);
  return a;
}

// B tile 32x16: B[kk,nn] = W[n0+nn, k0+kk], W row-major [Nout x K].
__device__ __forceinline__ v16h gB16(const _Float16* W, int ldK, int n0, int k0) {
  const int lane = threadIdx.x & 31;
  const _Float16* p = W + (size_t)(n0 + (lane & 15)) * ldK + k0 + ((lane >> 4) & 1) * 16;
  v16h b;
  *reinterpret_cast<uint4*>(&b)       = *reinterpret_cast<const uint4*>(p);
  *(reinterpret_cast<uint4*>(&b) + 1) = *reinterpret_cast<const uint4*>(p + 8);
  return b;
}

// fp32 fallback path (only if d_ws cannot hold f16 weights).
__device__ __forceinline__ v16h gB32(const float* W, int ldK, int n0, int k0) {
  const int lane = threadIdx.x & 31;
  const float* p = W + (size_t)(n0 + (lane & 15)) * ldK + k0 + ((lane >> 4) & 1) * 16;
  v16h b;
#pragma unroll
  for (int q = 0; q < 4; ++q) {
    float4 f = *reinterpret_cast<const float4*>(p + 4 * q);
    b[4*q+0] = (_Float16)f.x; b[4*q+1] = (_Float16)f.y;
    b[4*q+2] = (_Float16)f.z; b[4*q+3] = (_Float16)f.w;
  }
  return b;
}

__device__ __forceinline__ v8f wmma32(v16h a, v16h b, v8f c) {
  return __builtin_amdgcn_wmma_f32_16x16x32_f16(false, a, false, b, (short)0, c,
                                                false, false);
}

__device__ __forceinline__ void storeD(_Float16* Y, int strideH, int n0, v8f d,
                                       const float* bias, bool gelu) {
  const int lane = threadIdx.x & 31;
  const int col  = n0 + (lane & 15);
  const int rb   = ((lane >> 4) & 1) ? 8 : 0;
  const float bv = bias[col];
#pragma unroll
  for (int r = 0; r < 8; ++r) {
    float v = d[r] + bv;
    if (gelu) v = geluf(v);
    Y[(r + rb) * strideH + col] = (_Float16)v;
  }
}

__device__ __forceinline__ void storeDf(float* Y, int strideH, int n0, v8f d,
                                        const float* bias) {
  const int lane = threadIdx.x & 31;
  const int col  = n0 + (lane & 15);
  const int rb   = ((lane >> 4) & 1) ? 8 : 0;
  const float bv = bias[col];
#pragma unroll
  for (int r = 0; r < 8; ++r) Y[(r + rb) * strideH + col] = d[r] + bv;
}

// NT output tiles (cols n0 + t*nstep) sharing one A-tile load per k-step.
template <bool W16, int NT>
__device__ __forceinline__ void mmTileN(const _Float16* A, int As,
                                        const float* Wf, const _Float16* Wh,
                                        int ldK, int K, int n0, int nstep,
                                        const float* bias, bool gelu,
                                        _Float16* Y, int Ys) {
  v8f acc[NT];
#pragma unroll
  for (int t = 0; t < NT; ++t) acc[t] = (v8f){};
#pragma unroll 2
  for (int k0 = 0; k0 < K; k0 += 32) {
    const v16h a = ldsA(A, As, k0);
#pragma unroll
    for (int t = 0; t < NT; ++t) {
      v16h b = W16 ? gB16(Wh, ldK, n0 + t * nstep, k0)
                   : gB32(Wf, ldK, n0 + t * nstep, k0);
      acc[t] = wmma32(a, b, acc[t]);
    }
  }
#pragma unroll
  for (int t = 0; t < NT; ++t) storeD(Y, Ys, n0 + t * nstep, acc[t], bias, gelu);
}

// ---- fp32 -> f16 weight conversion into d_ws (rerun every launch) -----------
__global__ __launch_bounds__(256) void cvt_weights(Params p, _Float16* ws) {
  const float* srcs[8] = {p.obs_W, p.pre_W1, p.pre_W2, p.Wih,
                          p.Whh,   p.pk_W1,  p.pk_W2,  p.rt_W};
  const int offs[8] = {WS_OBSW, WS_PREW1, WS_PREW2, WS_WIH,
                       WS_WHH,  WS_PKW1,  WS_PKW2,  WS_RTW};
  const int ns[8]   = {8192, 163840, 32768, 49152, 49152, 131072, 32768, 65536};
  const int s = blockIdx.y;
  const float* src = srcs[s];
  _Float16* dst = ws + offs[s];
  const int n = ns[s];
  for (int i = blockIdx.x * 256 + threadIdx.x; i * 4 < n; i += gridDim.x * 256) {
    float4 f = *reinterpret_cast<const float4*>(src + i * 4);
    v4h o;
    o[0] = (_Float16)f.x; o[1] = (_Float16)f.y; o[2] = (_Float16)f.z; o[3] = (_Float16)f.w;
    *reinterpret_cast<v4h*>(dst + i * 4) = o;
  }
}

// ---- LDS layout (61568 B static, overlaid) ----------------------------------
//  MG @0 16x648 f16 | SCR @20736 (33280B overlaid) | PR @54016 16x136 f16
//  redS @58368 redQ @59392 | rowM/rowR @60416/60480 | TMP @60544 (256 f32)

template <bool W16>
__global__ __launch_bounds__(256) void node_core_kernel(Params p, const _Float16* ws) {
  __shared__ __align__(16) unsigned char smemRaw[61568];
  _Float16* MG   = (_Float16*)(smemRaw);
  unsigned char* scr = smemRaw + 20736;
  _Float16* OBSS = (_Float16*)(scr);
  _Float16* XN   = (_Float16*)(scr);
  _Float16* H1   = (_Float16*)(scr + 20736);
  _Float16* GI   = (_Float16*)(scr);
  _Float16* GH   = (_Float16*)(scr + 12544);
  _Float16* PKN  = (_Float16*)(scr);
  _Float16* RTN  = (_Float16*)(scr + 16640);
  _Float16* PK1  = (_Float16*)(scr + 16640);
  float*    PD   = (float*)(scr);
  _Float16* PR   = (_Float16*)(smemRaw + 54016);
  float* redS = (float*)(smemRaw + 58368);
  float* redQ = (float*)(smemRaw + 59392);
  float* rowM = (float*)(smemRaw + 60416);
  float* rowR = (float*)(smemRaw + 60480);
  float* TMP  = (float*)(smemRaw + 60544);

  const int  tid  = threadIdx.x;
  const int  wave = tid >> 5;
  const long t0   = (long)blockIdx.x * 16;
  // contiguous-chunk mapping for elementwise phases: row + 8 columns
  const int  er = tid & 15;            // row 0..15
  const int  eh = (tid >> 4) << 3;     // col base 0,8,...,120
  const long et = t0 + er;

  if (W16) {
    __builtin_prefetch(ws + WS_PREW1 + (size_t)tid * 64, 0, 1);
    __builtin_prefetch(ws + WS_PKW1  + (size_t)tid * 64, 0, 1);
  } else {
    __builtin_prefetch(p.pre_W1 + (size_t)tid * 64, 0, 1);
    __builtin_prefetch(p.pk_W1  + (size_t)tid * 64, 0, 1);
  }

  // ---- Phase 0: build merged[16x640] f16 in LDS (vectorized) ----
  {
    const int ni = p.nidx[et];
    float4 a0 = *(const float4*)(p.packet + et * HD + eh);
    float4 a1 = *(const float4*)(p.packet + et * HD + eh + 4);
    st8h(MG + er * 648 + eh, a0, a1);
    a0 = *(const float4*)(p.node + et * HD + eh);
    a1 = *(const float4*)(p.node + et * HD + eh + 4);
    st8h(MG + er * 648 + 128 + eh, a0, a1);
    a0 = *(const float4*)(p.embed + (size_t)ni * HD + eh);
    a1 = *(const float4*)(p.embed + (size_t)ni * HD + eh + 4);
    st8h(MG + er * 648 + 384 + eh, a0, a1);
    const float ag = p.age[et], tf = p.timef[et], rf = p.rem[et];
    const float nf = (float)ni * (1.0f / 2047.0f);
    v8h mv;
#pragma unroll
    for (int k = 0; k < 8; ++k) {
      float4 w = *(const float4*)(p.meta_W + (eh + k) * 4);
      mv[k] = (_Float16)(ag * w.x + tf * w.y + rf * w.z + nf * w.w + p.meta_b[eh + k]);
    }
    *reinterpret_cast<v8h*>(MG + er * 648 + 512 + eh) = mv;
    // obs staging [16x64]: 4 contiguous cols per thread
    const int cb = (tid >> 4) << 2;    // 0..60
    float4 o = *(const float4*)(p.obs + et * 64 + cb);
    v4h oh;
    oh[0] = (_Float16)o.x; oh[1] = (_Float16)o.y; oh[2] = (_Float16)o.z; oh[3] = (_Float16)o.w;
    *reinterpret_cast<v4h*>(OBSS + er * 72 + cb) = oh;
  }
  __syncthreads();
  // obs_f -> merged cols 256..383 (8 tiles / 8 waves)
  mmTileN<W16, 1>(OBSS, 72, p.obs_W, ws + WS_OBSW, 64, 64, wave * 16, 0,
                  p.obs_b, false, MG + 256, 648);
  __syncthreads();

  // ---- Phase 1: LN(merged, 640) -> XN ----
  {
    const int r = tid >> 4, c = tid & 15;      // 16 threads/row, 40-col chunks
    const _Float16* base = MG + r * 648 + c * 40;
    float s = 0.f, q = 0.f;
#pragma unroll
    for (int u = 0; u < 5; ++u) {
      v8h x = *reinterpret_cast<const v8h*>(base + u * 8);
#pragma unroll
      for (int k = 0; k < 8; ++k) { float v = (float)x[k]; s += v; q += v * v; }
    }
    redS[tid] = s; redQ[tid] = q;
  }
  __syncthreads();
  if (tid < 16) {
    float s = 0.f, q = 0.f;
#pragma unroll
    for (int c = 0; c < 16; ++c) { s += redS[tid * 16 + c]; q += redQ[tid * 16 + c]; }
    const float m = s * (1.0f / 640.0f);
    rowM[tid] = m; rowR[tid] = rsqrtf(q * (1.0f / 640.0f) - m * m + 1e-5f);
  }
  __syncthreads();
  {
    const int r = tid >> 4, c = tid & 15;
    const float m = rowM[r], rs = rowR[r];
    const _Float16* src = MG + r * 648 + c * 40;
    _Float16*       dst = XN + r * 648 + c * 40;
#pragma unroll
    for (int u = 0; u < 5; ++u) {
      v8h x = *reinterpret_cast<const v8h*>(src + u * 8);
      float4 g0 = *(const float4*)(p.pre_g  + c * 40 + u * 8);
      float4 g1 = *(const float4*)(p.pre_g  + c * 40 + u * 8 + 4);
      float4 b0 = *(const float4*)(p.pre_bln + c * 40 + u * 8);
      float4 b1 = *(const float4*)(p.pre_bln + c * 40 + u * 8 + 4);
      const float g[8] = {g0.x,g0.y,g0.z,g0.w,g1.x,g1.y,g1.z,g1.w};
      const float b[8] = {b0.x,b0.y,b0.z,b0.w,b1.x,b1.y,b1.z,b1.w};
      v8h y;
#pragma unroll
      for (int k = 0; k < 8; ++k)
        y[k] = (_Float16)(((float)x[k] - m) * rs * g[k] + b[k]);
      *reinterpret_cast<v8h*>(dst + u * 8) = y;
    }
  }
  __syncthreads();

  // ---- Phase 2: h1 = GELU(XN @ pre_W1.T + b1) [16x256], A shared x2 ----
  mmTileN<W16, 2>(XN, 648, p.pre_W1, ws + WS_PREW1, 640, 640, wave * 16, 128,
                  p.pre_b1, true, H1, 264);
  __syncthreads();

  // ---- Phase 3: proposal = h1 @ pre_W2.T + b2 [16x128] ----
  mmTileN<W16, 1>(H1, 264, p.pre_W2, ws + WS_PREW2, 256, 256, wave * 16, 0,
                  p.pre_b2, false, PR, 136);
  __syncthreads();

  // ---- Phase 4: GRU gates. Waves 0-3: gi (6 tiles, A=PR shared);
  //                          waves 4-7: gh (6 tiles, A=node cols of MG). ----
  if (wave < 4)
    mmTileN<W16, 6>(PR, 136, p.Wih, ws + WS_WIH, 128, 128, wave * 96, 16,
                    p.bih, false, GI, 392);
  else
    mmTileN<W16, 6>(MG + 128, 648, p.Whh, ws + WS_WHH, 128, 128, (wave - 4) * 96, 16,
                    p.bhh, false, GH, 392);
  __syncthreads();
  // GRU elementwise (vectorized): ns_next overwrites merged cols 128..255
  {
    v8h gi0 = *reinterpret_cast<const v8h*>(GI + er * 392 + eh);
    v8h gi1 = *reinterpret_cast<const v8h*>(GI + er * 392 + 128 + eh);
    v8h gi2 = *reinterpret_cast<const v8h*>(GI + er * 392 + 256 + eh);
    v8h gh0 = *reinterpret_cast<const v8h*>(GH + er * 392 + eh);
    v8h gh1 = *reinterpret_cast<const v8h*>(GH + er * 392 + 128 + eh);
    v8h gh2 = *reinterpret_cast<const v8h*>(GH + er * 392 + 256 + eh);
    float4 n0 = *(const float4*)(p.node + et * HD + eh);
    float4 n1 = *(const float4*)(p.node + et * HD + eh + 4);
    const float nsold[8] = {n0.x,n0.y,n0.z,n0.w,n1.x,n1.y,n1.z,n1.w};
    float ns[8]; v8h nh;
#pragma unroll
    for (int k = 0; k < 8; ++k) {
      const float rr = 1.f / (1.f + expf(-((float)gi0[k] + (float)gh0[k])));
      const float zz = 1.f / (1.f + expf(-((float)gi1[k] + (float)gh1[k])));
      const float nn = tanhf((float)gi2[k] + rr * (float)gh2[k]);
      ns[k] = (1.f - zz) * nn + zz * nsold[k];
      nh[k] = (_Float16)ns[k];
    }
    float4 o0 = make_float4(ns[0], ns[1], ns[2], ns[3]);
    float4 o1 = make_float4(ns[4], ns[5], ns[6], ns[7]);
    *(float4*)(p.out_ns + et * HD + eh)     = o0;
    *(float4*)(p.out_ns + et * HD + eh + 4) = o1;
    *reinterpret_cast<v8h*>(MG + er * 648 + 128 + eh) = nh;
  }
  __syncthreads();

  // ---- Phase 5: LN(pf=merged[:,0:512]) -> PKN (pk_ln) and RTN (rt_ln) ----
  {
    const int r = tid >> 4, c = tid & 15;      // 32-col chunks
    const _Float16* base = MG + r * 648 + c * 32;
    float s = 0.f, q = 0.f;
#pragma unroll
    for (int u = 0; u < 4; ++u) {
      v8h x = *reinterpret_cast<const v8h*>(base + u * 8);
#pragma unroll
      for (int k = 0; k < 8; ++k) { float v = (float)x[k]; s += v; q += v * v; }
    }
    redS[tid] = s; redQ[tid] = q;
  }
  __syncthreads();
  if (tid < 16) {
    float s = 0.f, q = 0.f;
#pragma unroll
    for (int c = 0; c < 16; ++c) { s += redS[tid * 16 + c]; q += redQ[tid * 16 + c]; }
    const float m = s * (1.0f / 512.0f);
    rowM[tid] = m; rowR[tid] = rsqrtf(q * (1.0f / 512.0f) - m * m + 1e-5f);
  }
  __syncthreads();
  {
    const int r = tid >> 4, c = tid & 15;
    const float m = rowM[r], rs = rowR[r];
    const _Float16* src = MG + r * 648 + c * 32;
#pragma unroll
    for (int u = 0; u < 4; ++u) {
      v8h x = *reinterpret_cast<const v8h*>(src + u * 8);
      const int jb = c * 32 + u * 8;
      float4 pg0 = *(const float4*)(p.pk_g + jb),   pg1 = *(const float4*)(p.pk_g + jb + 4);
      float4 pb0 = *(const float4*)(p.pk_bln + jb), pb1 = *(const float4*)(p.pk_bln + jb + 4);
      float4 rg0 = *(const float4*)(p.rt_g + jb),   rg1 = *(const float4*)(p.rt_g + jb + 4);
      float4 rb0 = *(const float4*)(p.rt_bln + jb), rb1 = *(const float4*)(p.rt_bln + jb + 4);
      const float pg[8] = {pg0.x,pg0.y,pg0.z,pg0.w,pg1.x,pg1.y,pg1.z,pg1.w};
      const float pb[8] = {pb0.x,pb0.y,pb0.z,pb0.w,pb1.x,pb1.y,pb1.z,pb1.w};
      const float rg[8] = {rg0.x,rg0.y,rg0.z,rg0.w,rg1.x,rg1.y,rg1.z,rg1.w};
      const float rb[8] = {rb0.x,rb0.y,rb0.z,rb0.w,rb1.x,rb1.y,rb1.z,rb1.w};
      v8h yp, yr;
#pragma unroll
      for (int k = 0; k < 8; ++k) {
        const float nv = ((float)x[k] - m) * rs;
        yp[k] = (_Float16)(nv * pg[k] + pb[k]);
        yr[k] = (_Float16)(nv * rg[k] + rb[k]);
      }
      *reinterpret_cast<v8h*>(PKN + r * 520 + jb) = yp;
      *reinterpret_cast<v8h*>(RTN + r * 520 + jb) = yr;
    }
  }
  __syncthreads();

  // ---- Phase 6: router hidden rh = GELU(RTN @ rt_W.T + rt_b) ----
  mmTileN<W16, 1>(RTN, 520, p.rt_W, ws + WS_RTW, 512, 512, wave * 16, 0,
                  p.rt_b, true, PR, 136);
  __syncthreads();

  // ---- Phase 7: pk1 = GELU(PKN @ pk_W1.T + b1) [16x256], A shared x2 ----
  mmTileN<W16, 2>(PKN, 520, p.pk_W1, ws + WS_PKW1, 512, 512, wave * 16, 128,
                  p.pk_b1, true, PK1, 264);
  __syncthreads();

  // ---- Phase 8: pd = pk1 @ pk_W2.T + b2 [16x128] f32 (over PKN) ----
  {
    v8f acc = {};
#pragma unroll
    for (int k0 = 0; k0 < 256; k0 += 32) {
      v16h b = W16 ? gB16(ws + WS_PKW2, 256, wave * 16, k0)
                   : gB32(p.pk_W2, 256, wave * 16, k0);
      acc = wmma32(ldsA(PK1, 264, k0), b, acc);
    }
    storeDf(PD, 132, wave * 16, acc, p.pk_b2);
  }
  __syncthreads();

  // ---- Phase 9: rank-16 adapter: tmp[r,j] = sum_h pd[r,h]*ad_down[j,h] ----
  {
    const int r = tid >> 4, j = tid & 15;
    float s = 0.f;
#pragma unroll 8
    for (int h = 0; h < HD; h += 4) {
      float4 dv = *(const float4*)(PD + r * 132 + h);
      float4 wv = *(const float4*)(p.ad_down + j * HD + h);
      s += dv.x * wv.x + dv.y * wv.y + dv.z * wv.z + dv.w * wv.w;
    }
    TMP[tid] = s;
  }
  __syncthreads();

  // ---- Phase 10: packet_next (vectorized) + router logits ----
  {
    float4 d0 = *(const float4*)(PD + er * 132 + eh);
    float4 d1 = *(const float4*)(PD + er * 132 + eh + 4);
    float4 p0 = *(const float4*)(p.packet + et * HD + eh);
    float4 p1 = *(const float4*)(p.packet + et * HD + eh + 4);
    float tj[16];
#pragma unroll
    for (int u = 0; u < 4; ++u) {
      float4 tv = *(const float4*)(TMP + er * 16 + u * 4);
      tj[4*u] = tv.x; tj[4*u+1] = tv.y; tj[4*u+2] = tv.z; tj[4*u+3] = tv.w;
    }
    const float dd[8] = {d0.x,d0.y,d0.z,d0.w,d1.x,d1.y,d1.z,d1.w};
    const float pp[8] = {p0.x,p0.y,p0.z,p0.w,p1.x,p1.y,p1.z,p1.w};
    float ov[8];
#pragma unroll
    for (int k = 0; k < 8; ++k) {
      float v = dd[k];
#pragma unroll
      for (int u = 0; u < 4; ++u) {
        float4 uw = *(const float4*)(p.ad_up + (eh + k) * 16 + u * 4);
        v += tj[4*u] * uw.x + tj[4*u+1] * uw.y + tj[4*u+2] * uw.z + tj[4*u+3] * uw.w;
      }
      ov[k] = pp[k] + v;
    }
    *(float4*)(p.out_pk + et * HD + eh)     = make_float4(ov[0], ov[1], ov[2], ov[3]);
    *(float4*)(p.out_pk + et * HD + eh + 4) = make_float4(ov[4], ov[5], ov[6], ov[7]);
  }
  if (tid < 48) {
    const int r = tid / 3, o = tid - 3 * (tid / 3);
    float s = p.ro_b[o];
#pragma unroll
    for (int h = 0; h < HD; h += 8) {
      v8h x = *reinterpret_cast<const v8h*>(PR + r * 136 + h);
      float4 w0 = *(const float4*)(p.ro_W + o * HD + h);
      float4 w1 = *(const float4*)(p.ro_W + o * HD + h + 4);
      s += (float)x[0]*w0.x + (float)x[1]*w0.y + (float)x[2]*w0.z + (float)x[3]*w0.w
         + (float)x[4]*w1.x + (float)x[5]*w1.y + (float)x[6]*w1.z + (float)x[7]*w1.w;
    }
    p.out_lg[(t0 + r) * 3 + o] = s;
  }
}

extern "C" void kernel_launch(void* const* d_in, const int* in_sizes, int n_in,
                              void* d_out, int out_size, void* d_ws, size_t ws_size,
                              hipStream_t stream) {
  (void)in_sizes; (void)n_in; (void)out_size;
  Params p;
  p.packet  = (const float*)d_in[0];
  p.node    = (const float*)d_in[1];
  p.obs     = (const float*)d_in[2];
  p.age     = (const float*)d_in[3];
  p.timef   = (const float*)d_in[4];
  p.rem     = (const float*)d_in[5];
  p.nidx    = (const int*)  d_in[6];
  p.obs_W   = (const float*)d_in[7];
  p.obs_b   = (const float*)d_in[8];
  p.meta_W  = (const float*)d_in[9];
  p.meta_b  = (const float*)d_in[10];
  p.embed   = (const float*)d_in[11];
  p.pre_g   = (const float*)d_in[12];
  p.pre_bln = (const float*)d_in[13];
  p.pre_W1  = (const float*)d_in[14];
  p.pre_b1  = (const float*)d_in[15];
  p.pre_W2  = (const float*)d_in[16];
  p.pre_b2  = (const float*)d_in[17];
  p.Wih     = (const float*)d_in[18];
  p.Whh     = (const float*)d_in[19];
  p.bih     = (const float*)d_in[20];
  p.bhh     = (const float*)d_in[21];
  p.pk_g    = (const float*)d_in[22];
  p.pk_bln  = (const float*)d_in[23];
  p.pk_W1   = (const float*)d_in[24];
  p.pk_b1   = (const float*)d_in[25];
  p.pk_W2   = (const float*)d_in[26];
  p.pk_b2   = (const float*)d_in[27];
  p.ad_down = (const float*)d_in[28];
  p.ad_up   = (const float*)d_in[29];
  p.rt_g    = (const float*)d_in[30];
  p.rt_bln  = (const float*)d_in[31];
  p.rt_W    = (const float*)d_in[32];
  p.rt_b    = (const float*)d_in[33];
  p.ro_W    = (const float*)d_in[34];
  p.ro_b    = (const float*)d_in[35];
  float* out = (float*)d_out;
  p.out_ns = out;
  p.out_pk = out + (size_t)NTOK * HD;
  p.out_lg = out + (size_t)2 * NTOK * HD;

  const size_t needWs = (size_t)WS_TOTAL * sizeof(_Float16);
  if (d_ws != nullptr && ws_size >= needWs) {
    _Float16* ws = (_Float16*)d_ws;
    cvt_weights<<<dim3(64, 8), dim3(256), 0, stream>>>(p, ws);
    node_core_kernel<true><<<dim3(NTOK / 16), dim3(256), 0, stream>>>(p, ws);
  } else {
    node_core_kernel<false><<<dim3(NTOK / 16), dim3(256), 0, stream>>>(
        p, (const _Float16*)nullptr);
  }
}
